// MultiEncoderLayer_58720792871161
// MI455X (gfx1250) — compile-verified, matrix operand order checked by reading, not run on hardware
//
#include <hip/hip_runtime.h>
#include <hip/hip_bf16.h>

typedef __bf16 bf16;
typedef __attribute__((ext_vector_type(16))) __bf16 v16bf;
typedef __attribute__((ext_vector_type(8)))  __bf16 v8bf;
typedef __attribute__((ext_vector_type(8)))  float  v8f;
typedef __attribute__((ext_vector_type(4)))  unsigned u32x4;
typedef __attribute__((ext_vector_type(8)))  unsigned u32x8;

#define ZERO8 {0.f,0.f,0.f,0.f,0.f,0.f,0.f,0.f}

// ---------------------------------------------------------------------------
// WMMA helper (CDNA5 gfx1250, wave32)
// ---------------------------------------------------------------------------
__device__ __forceinline__ v8f wmma_bf16(v16bf a, v16bf b, v8f c) {
  return __builtin_amdgcn_wmma_f32_16x16x32_bf16(false, a, false, b, (short)0, c,
                                                 false, false);
}

// Load a 16x32 bf16 A-fragment from a row-major tile (row stride ld elements).
// ISA layout: lanes 0-15 (rows 0-15) hold K = {0..7, 16..23};
//             lanes 16-31 (rows 0-15) hold K = {8..15, 24..31}.
// Same loader serves the B operand when `base` points at the row-major
// N x K (B-transposed) matrix. Works for global or LDS-backed pointers.
__device__ __forceinline__ v16bf load_frag(const bf16* base, int ld, int lane) {
  int m  = lane & 15;
  int kb = (lane & 16) ? 8 : 0;
  const bf16* p = base + (size_t)m * ld + kb;
  union { v16bf v; v8bf h[2]; } u;
  u.h[0] = *(const v8bf*)(p);
  u.h[1] = *(const v8bf*)(p + 16);
  return u.v;
}

// ---------------------------------------------------------------------------
// Tensor Data Mover: stage a 2D bf16 tile (rows x width, row pitch `stride`)
// from global memory into LDS (packed rows). Issued by ONE wave; caller must
// s_wait_tensorcnt + barrier before other waves consume the LDS tile.
// D# layout per CDNA5 ISA.
// ---------------------------------------------------------------------------
__device__ __forceinline__ void tdm_load_tile_b16(unsigned lds_off,
                                                  const void* gptr,
                                                  unsigned width, unsigned rows,
                                                  unsigned stride) {
  unsigned long long ga = (unsigned long long)gptr;
  union { u32x4 v; unsigned u[4]; } g0;
  g0.u[0] = 1u;                                   // count=1, user descriptor
  g0.u[1] = lds_off;                              // lds_addr (bytes)
  g0.u[2] = (unsigned)ga;                         // global_addr[31:0]
  g0.u[3] = (unsigned)((ga >> 32) & 0x01FFFFFFu)  // global_addr[56:32]
          | 0x80000000u;                          // type=2 ("image")
  union { u32x8 v; unsigned u[8]; } g1;
  g1.u[0] = (1u << 16);                           // wg_mask=0, data_size=1 (2B)
  g1.u[1] = (width & 0xFFFFu) << 16;              // tensor_dim0[15:0] @ bit48
  g1.u[2] = (width >> 16) | ((rows & 0xFFFFu) << 16); // dim0[31:16], dim1[15:0]
  g1.u[3] = (rows >> 16) | ((width & 0xFFFFu) << 16); // dim1[31:16], tile_dim0
  g1.u[4] = rows & 0xFFFFu;                       // tile_dim1, tile_dim2=0
  g1.u[5] = stride;                               // tensor_dim0_stride[31:0]
  g1.u[6] = 0u;                                   // stride[47:32]=0, dim1_stride=0
  g1.u[7] = 0u;
  asm volatile("tensor_load_to_lds %0, %1" :: "s"(g0.v), "s"(g1.v) : "memory");
}

__device__ __forceinline__ unsigned lds_offset_of(const void* p) {
  // Generic pointers to LDS carry the byte offset in the low 32 bits.
  return (unsigned)(unsigned long long)p;
}

// ---------------------------------------------------------------------------
// K1: per-batch max over [L*L] for t_m (blockIdx.y==0) and g_m (blockIdx.y==1)
// ---------------------------------------------------------------------------
__global__ __launch_bounds__(256)
void bmax_kernel(const float* __restrict__ t, const float* __restrict__ g,
                 float* __restrict__ outmax, int B, long long N) {
  int b = blockIdx.x, which = blockIdx.y, tid = threadIdx.x;
  const float* src = (which == 0 ? t : g) + (size_t)b * N;
  float mx = -3.0e38f;
  for (long long i = tid; i < N; i += blockDim.x) mx = fmaxf(mx, src[i]);
  __shared__ float red[256];
  red[tid] = mx; __syncthreads();
  for (int off = 128; off > 0; off >>= 1) {
    if (tid < off) red[tid] = fmaxf(red[tid], red[tid + off]);
    __syncthreads();
  }
  if (tid == 0) outmax[which * B + b] = red[0];
}

// ---------------------------------------------------------------------------
// K2/K6: layernorm of one row (grid = B*L), write bf16 h and (optionally) h^T
// ---------------------------------------------------------------------------
__global__ __launch_bounds__(128)
void ln_kernel(const float* __restrict__ x, const float* __restrict__ w,
               const float* __restrict__ bia, bf16* __restrict__ h,
               bf16* __restrict__ hT, int L, int D) {
  int row = blockIdx.x;               // b*L + l
  int bidx = row / L, l = row % L, tid = threadIdx.x;
  const float* xr = x + (size_t)row * D;
  float s = 0.f, s2 = 0.f;
  for (int d = tid; d < D; d += blockDim.x) { float v = xr[d]; s += v; s2 += v * v; }
  __shared__ float r1[128], r2[128];
  r1[tid] = s; r2[tid] = s2; __syncthreads();
  for (int off = 64; off > 0; off >>= 1) {
    if (tid < off) { r1[tid] += r1[tid + off]; r2[tid] += r2[tid + off]; }
    __syncthreads();
  }
  float mu  = r1[0] / (float)D;
  float var = r2[0] / (float)D - mu * mu;
  float inv = rsqrtf(var + 1e-6f);
  for (int d = tid; d < D; d += blockDim.x) {
    float v = (xr[d] - mu) * inv * w[d] + bia[d];
    bf16 hv = (bf16)v;
    h[(size_t)row * D + d] = hv;
    if (hT) hT[((size_t)bidx * D + d) * L + l] = hv;
  }
}

// ---------------------------------------------------------------------------
// K3: convert + transpose weights once:  w1[D,F]->w1T[F,D], w2[F,D]->w2T[D,F]
// ---------------------------------------------------------------------------
__global__ __launch_bounds__(256)
void wconv_kernel(const float* __restrict__ w1, const float* __restrict__ w2,
                  bf16* __restrict__ w1T, bf16* __restrict__ w2T, int D, int F) {
  size_t n = (size_t)D * F;
  for (size_t i = (size_t)blockIdx.x * blockDim.x + threadIdx.x; i < n;
       i += (size_t)gridDim.x * blockDim.x) {
    size_t d = i / F, f = i % F;
    w1T[f * D + d] = (bf16)w1[i];
    size_t f2 = i / D, d2 = i % D;
    w2T[d2 * F + f2] = (bf16)w2[i];
  }
}

// ---------------------------------------------------------------------------
// K4: scores + bias + mask + softmax -> P (bf16).
//     4 waves per block share one 16x1024 f32 score strip (64KB LDS):
//     wave w computes k-tiles w, w+4, ..., then softmaxes rows 4w..4w+3.
// ---------------------------------------------------------------------------
__global__ __launch_bounds__(128)
void attn_scores_kernel(const bf16* __restrict__ h, const float* __restrict__ t_m,
                        const float* __restrict__ g_m, const int* __restrict__ mask,
                        const float* __restrict__ maxes, bf16* __restrict__ P,
                        int B, int L, int D) {
  __shared__ float S[16][1024];                       // 64 KB
  int lane = threadIdx.x & 31, wave = threadIdx.x >> 5;
  int b = blockIdx.y, qbase = blockIdx.x * 16;
  const bf16* hb = h + (size_t)b * L * D;
  float tmax = maxes[b], gmax = maxes[B + b];
  const float invd = 1.0f / (float)D;
  int n = lane & 15;
  int roff = (lane & 16) ? 8 : 0;

  // Preload the 16 A-fragments of the q-tile (K = D = 512 -> 16 chunks of 32).
  v16bf afr[16];
  for (int dc = 0; dc < 16; ++dc)
    afr[dc] = load_frag(hb + (size_t)qbase * D + dc * 32, D, lane);

  for (int kt = wave; kt < L / 16; kt += 4) {
    v8f acc = ZERO8;
    for (int dc = 0; dc < 16; ++dc) {
      v16bf bm = load_frag(hb + (size_t)(kt * 16) * D + dc * 32, D, lane);
      acc = wmma_bf16(afr[dc], bm, acc);
    }
    int kg = kt * 16 + n;
    for (int r = 0; r < 8; ++r) {
      int qg = qbase + r + roff;
      size_t off = ((size_t)b * L + qg) * L + kg;
      float s = acc[r] * invd + fabsf(t_m[off] - tmax) + fabsf(g_m[off] - gmax);
      if (mask[off] == 0) s = -1000000000.0f;
      S[r + roff][kg] = s;
    }
  }
  __syncthreads();

  // Row softmax: each wave owns 4 rows; 32 lanes cooperate per row.
  for (int row = wave * 4; row < wave * 4 + 4; ++row) {
    float mx = -3.0e38f;
    for (int c = lane; c < L; c += 32) mx = fmaxf(mx, S[row][c]);
    for (int o = 16; o > 0; o >>= 1) mx = fmaxf(mx, __shfl_xor(mx, o, 32));
    float sm = 0.f;
    for (int c = lane; c < L; c += 32) {
      float e = __expf(S[row][c] - mx); S[row][c] = e; sm += e;
    }
    for (int o = 16; o > 0; o >>= 1) sm += __shfl_xor(sm, o, 32);
    float inv = 1.0f / sm;
    bf16* Pr = P + ((size_t)b * L + qbase + row) * L;
    for (int c = lane; c < L; c += 32) Pr[c] = (bf16)(S[row][c] * inv);
  }
}

// ---------------------------------------------------------------------------
// K5: O = P @ h (via h^T), fused residual x1 = x + O.
//     P q-tile (16x1024 bf16, 32KB) staged into LDS once per block by the TDM;
//     8 waves each own 64 of the 512 output columns.
// ---------------------------------------------------------------------------
__global__ __launch_bounds__(256)
void attn_pv_kernel(const bf16* __restrict__ P, const bf16* __restrict__ hT,
                    const float* __restrict__ x, float* __restrict__ x1,
                    int L, int D) {
  __shared__ bf16 sP[16 * 1024];                      // 32 KB
  int lane = threadIdx.x & 31, wave = threadIdx.x >> 5;
  int b = blockIdx.y, qbase = blockIdx.x * 16;
  int n0 = wave * 64;                                 // 8 waves * 64 = D
  const bf16* Pb  = P  + ((size_t)b * L + qbase) * L;
  const bf16* hTb = hT + (size_t)b * D * L;

  if (threadIdx.x < 32) {                             // wave 0 drives the TDM
    tdm_load_tile_b16(lds_offset_of(sP), Pb, (unsigned)L, 16u, (unsigned)L);
    __builtin_amdgcn_s_wait_tensorcnt(0);
  }
  __syncthreads();

  const v8f vz = ZERO8;
  v8f acc[4] = {vz, vz, vz, vz};
  for (int kc = 0; kc < L / 32; ++kc) {
    v16bf a = load_frag(sP + kc * 32, L, lane);       // ds_load from LDS
    for (int t = 0; t < 4; ++t) {
      v16bf bm = load_frag(hTb + (size_t)(n0 + t * 16) * L + kc * 32, L, lane);
      acc[t] = wmma_bf16(a, bm, acc[t]);
    }
  }
  int roff = (lane & 16) ? 8 : 0, n = lane & 15;
  for (int t = 0; t < 4; ++t)
    for (int r = 0; r < 8; ++r) {
      int qg = qbase + r + roff, d = n0 + t * 16 + n;
      size_t idx = ((size_t)b * L + qg) * D + d;
      x1[idx] = x[idx] + acc[t][r];
    }
}

// ---------------------------------------------------------------------------
// K7: act = relu(h2 @ w1 + b1) -> bf16.
//     h2 q-tile (16x512 bf16, 16KB) staged into LDS via TDM; each wave does
//     16 rows x 32 cols.
// ---------------------------------------------------------------------------
__global__ __launch_bounds__(256)
void ffn1_kernel(const bf16* __restrict__ h2, const bf16* __restrict__ w1T,
                 const float* __restrict__ b1, bf16* __restrict__ act,
                 int L, int D, int F) {
  __shared__ bf16 sA[16 * 512];                       // 16 KB
  int lane = threadIdx.x & 31, wave = threadIdx.x >> 5;
  int nq = L / 16;
  int b = blockIdx.x / nq, qbase = (blockIdx.x % nq) * 16;
  int n0 = blockIdx.y * 256 + wave * 32;
  const bf16* A = h2 + ((size_t)b * L + qbase) * D;

  if (threadIdx.x < 32) {
    tdm_load_tile_b16(lds_offset_of(sA), A, (unsigned)D, 16u, (unsigned)D);
    __builtin_amdgcn_s_wait_tensorcnt(0);
  }
  __syncthreads();

  const v8f vz = ZERO8;
  v8f acc[2] = {vz, vz};
  for (int kc = 0; kc < D / 32; ++kc) {
    v16bf a = load_frag(sA + kc * 32, D, lane);
    for (int t = 0; t < 2; ++t) {
      v16bf bm = load_frag(w1T + (size_t)(n0 + t * 16) * D + kc * 32, D, lane);
      acc[t] = wmma_bf16(a, bm, acc[t]);
    }
  }
  int roff = (lane & 16) ? 8 : 0, n = lane & 15;
  for (int t = 0; t < 2; ++t)
    for (int r = 0; r < 8; ++r) {
      int qg = qbase + r + roff, f = n0 + t * 16 + n;
      float v = fmaxf(acc[t][r] + b1[f], 0.0f);
      act[((size_t)b * L + qg) * F + f] = (bf16)v;
    }
}

// ---------------------------------------------------------------------------
// K8: out = (x1 + act @ w2 + b2) / D.
//     act q-tile (16x2048 bf16, 64KB) staged into LDS via TDM.
// ---------------------------------------------------------------------------
__global__ __launch_bounds__(256)
void ffn2_kernel(const bf16* __restrict__ act, const bf16* __restrict__ w2T,
                 const float* __restrict__ b2, const float* __restrict__ x1,
                 float* __restrict__ out, int L, int D, int F) {
  __shared__ bf16 sA[16 * 2048];                      // 64 KB
  int lane = threadIdx.x & 31, wave = threadIdx.x >> 5;
  int nq = L / 16;
  int b = blockIdx.x / nq, qbase = (blockIdx.x % nq) * 16;
  int n0 = blockIdx.y * 256 + wave * 32;
  const bf16* A = act + ((size_t)b * L + qbase) * F;

  if (threadIdx.x < 32) {
    tdm_load_tile_b16(lds_offset_of(sA), A, (unsigned)F, 16u, (unsigned)F);
    __builtin_amdgcn_s_wait_tensorcnt(0);
  }
  __syncthreads();

  const v8f vz = ZERO8;
  v8f acc[2] = {vz, vz};
  for (int kc = 0; kc < F / 32; ++kc) {
    v16bf a = load_frag(sA + kc * 32, F, lane);
    for (int t = 0; t < 2; ++t) {
      v16bf bm = load_frag(w2T + (size_t)(n0 + t * 16) * F + kc * 32, F, lane);
      acc[t] = wmma_bf16(a, bm, acc[t]);
    }
  }
  int roff = (lane & 16) ? 8 : 0, n = lane & 15;
  float invd = 1.0f / (float)D;
  for (int t = 0; t < 2; ++t)
    for (int r = 0; r < 8; ++r) {
      int qg = qbase + r + roff, d = n0 + t * 16 + n;
      size_t idx = ((size_t)b * L + qg) * D + d;
      out[idx] = (x1[idx] + acc[t][r] + b2[d]) * invd;
    }
}

// ---------------------------------------------------------------------------
extern "C" void kernel_launch(void* const* d_in, const int* in_sizes, int n_in,
                              void* d_out, int out_size, void* d_ws, size_t ws_size,
                              hipStream_t stream) {
  const int B = 16, L = 1024, D = 512, F = 2048;
  const float* x    = (const float*)d_in[0];
  const float* t_m  = (const float*)d_in[1];
  const float* g_m  = (const float*)d_in[2];
  const int*   mask = (const int*)d_in[3];
  const float* ln1w = (const float*)d_in[4];
  const float* ln1b = (const float*)d_in[5];
  const float* ln2w = (const float*)d_in[6];
  const float* ln2b = (const float*)d_in[7];
  const float* w1   = (const float*)d_in[8];
  const float* b1   = (const float*)d_in[9];
  const float* w2   = (const float*)d_in[10];
  const float* b2   = (const float*)d_in[11];
  float* out = (float*)d_out;

  char* ws = (char*)d_ws;
  auto alloc = [&](size_t bytes) {
    char* p = ws;
    ws += (bytes + 255) & ~(size_t)255;
    return p;
  };
  float* maxes = (float*)alloc(2 * B * sizeof(float));
  bf16*  h     = (bf16*) alloc((size_t)B * L * D * sizeof(bf16));
  bf16*  hT    = (bf16*) alloc((size_t)B * L * D * sizeof(bf16));
  bf16*  h2    = (bf16*) alloc((size_t)B * L * D * sizeof(bf16));
  bf16*  P     = (bf16*) alloc((size_t)B * L * L * sizeof(bf16));
  float* x1    = (float*)alloc((size_t)B * L * D * sizeof(float));
  bf16*  act   = (bf16*) alloc((size_t)B * L * F * sizeof(bf16));
  bf16*  w1T   = (bf16*) alloc((size_t)D * F * sizeof(bf16));
  bf16*  w2T   = (bf16*) alloc((size_t)D * F * sizeof(bf16));

  bmax_kernel<<<dim3(B, 2), 256, 0, stream>>>(t_m, g_m, maxes, B, (long long)L * L);
  ln_kernel<<<B * L, 128, 0, stream>>>(x, ln1w, ln1b, h, hT, L, D);
  wconv_kernel<<<2048, 256, 0, stream>>>(w1, w2, w1T, w2T, D, F);
  attn_scores_kernel<<<dim3(L / 16, B), 128, 0, stream>>>(h, t_m, g_m, mask, maxes,
                                                          P, B, L, D);
  attn_pv_kernel<<<dim3(L / 16, B), 256, 0, stream>>>(P, hT, x, x1, L, D);
  ln_kernel<<<B * L, 128, 0, stream>>>(x1, ln2w, ln2b, h2, nullptr, L, D);
  ffn1_kernel<<<dim3(B * (L / 16), F / 256), 256, 0, stream>>>(h2, w1T, b1, act,
                                                               L, D, F);
  ffn2_kernel<<<dim3(B * (L / 16), D / 256), 256, 0, stream>>>(act, w2T, b2, x1,
                                                               out, L, D, F);
}
